// LstmDecoder_86964497809446
// MI455X (gfx1250) — matrix operand order repaired and checked by reading, not compiled, not run on hardware
//
#include <hip/hip_runtime.h>
#include <hip/hip_bf16.h>
#include <stdint.h>

typedef __bf16 bf16;
typedef __attribute__((ext_vector_type(16))) __bf16 v16bf;
typedef __attribute__((ext_vector_type(8)))  __bf16 v8bf;
typedef __attribute__((ext_vector_type(8)))  float   v8f;

#define HID   1024
#define BATCH 256
#define OUTD  512
#define SEQ   128

// gemm kernel staging: 32 rows, K-phase 512, padded stride (260 dwords % 64 = 4 banks)
#define KPH_G    512
#define STRIDE_G 520
// lstm step staging: 64 rows, K-phase 256, padded stride (132 dwords % 64 = 4 banks)
#define KPH_S    256
#define STRIDE_S 264

__device__ __forceinline__ float sigf(float x) { return 1.0f / (1.0f + __expf(-x)); }

// ---------------- small helper kernels ----------------

__global__ __launch_bounds__(256) void cvt_kernel(const float* __restrict__ s,
                                                  bf16* __restrict__ d, int n) {
  int i = blockIdx.x * blockDim.x + threadIdx.x;
  int stride = gridDim.x * blockDim.x;
  for (; i < n; i += stride) d[i] = (bf16)s[i];
}

__global__ __launch_bounds__(256) void bias_sum_kernel(const float* __restrict__ a,
                                                       const float* __restrict__ b,
                                                       float* __restrict__ d, int n) {
  int i = blockIdx.x * blockDim.x + threadIdx.x;
  if (i < n) d[i] = a[i] + b[i];
}

__global__ __launch_bounds__(256) void zero_state_kernel(float* __restrict__ c,
                                                         bf16* __restrict__ h,
                                                         int nc, int nh) {
  int i = blockIdx.x * blockDim.x + threadIdx.x;
  int stride = gridDim.x * blockDim.x;
  for (int j = i; j < nc; j += stride) c[j] = 0.0f;
  for (int j = i; j < nh; j += stride) h[j] = (bf16)0.0f;
}

// ---------------- generic bf16 WMMA GEMM: out[m][n] = sum_k A[m][k]*W[n][k] + bias[n] ----
// block = 256 threads (8 waves), tile BM=32 x BN=64; barrier-free K loop.

__global__ __launch_bounds__(256)
void gemm_bf16_kernel(const bf16* __restrict__ A,
                      const bf16* __restrict__ W,
                      const float* __restrict__ bias,
                      float* __restrict__ out, int N) {
  __shared__ __align__(16) bf16 lds_a[32 * STRIDE_G];   // ~33 KB
  const int tid  = threadIdx.x;
  const int wave = tid >> 5;
  const int lane = tid & 31;
  const int nl   = lane & 15;
  const int hi   = lane >> 4;
  const int mbase0 = blockIdx.y * 32;
  const int mbase  = mbase0 + (wave >> 2) * 16;
  const int nbase  = blockIdx.x * 64 + (wave & 3) * 16;
  v8f acc = {};
  const bf16* wr  = W + (size_t)(nbase + nl) * HID;
  const int arow  = (wave >> 2) * 16 + nl;
  const int aoff  = hi * 8;

  for (int p = 0; p < 2; ++p) {
    const int kp = p * KPH_G;
    if (p) __syncthreads();
#pragma unroll
    for (int i = 0; i < 8; ++i) {           // 2048 chunks / 256 threads
      const int c   = tid + i * 256;
      const int row = c >> 6;               // 64 chunks per row
      const int col = (c & 63) * 8;
      *(v8bf*)&lds_a[row * STRIDE_G + col] =
          *(const v8bf*)&A[(size_t)(mbase0 + row) * HID + kp + col];
    }
    __syncthreads();

    for (int k0 = 0; k0 < KPH_G; k0 += 32) {
      v16bf af;
      ((v8bf*)&af)[0] = *(const v8bf*)&lds_a[arow * STRIDE_G + k0 + aoff];
      ((v8bf*)&af)[1] = *(const v8bf*)&lds_a[arow * STRIDE_G + k0 + aoff + 16];
      const bf16* bp = wr + kp + k0 + hi * 8;
      v16bf bfg;
      ((v8bf*)&bfg)[0] = *(const v8bf*)bp;
      ((v8bf*)&bfg)[1] = *(const v8bf*)(bp + 16);
      __builtin_prefetch(bp + 64, 0, 3);
      acc = __builtin_amdgcn_wmma_f32_16x16x32_bf16(false, af, false, bfg,
                                                    (short)0, acc, false, false);
    }
  }

  const int n    = nbase + nl;
  const float b  = bias[n];
  const int mloc = hi * 8;
#pragma unroll
  for (int v = 0; v < 8; ++v)
    out[(size_t)(mbase + mloc + v) * N + n] = acc[v] + b;
}

// ---------------- fused LSTM step -----------------------------------------------------
// block = 128 threads (4 waves), tile BM=64 x BN=32. Each wave: 2 m-positions x 4 gates
// = 8 accumulators; every B fragment (W_hh rows) feeds two WMMAs from registers.
// h tile (64 rows x 256-K phase) staged via gfx1250 async-to-LDS; K loop barrier-free.

__global__ __launch_bounds__(128)
void lstm_step_kernel(const float* __restrict__ xproj,   // [256 x 4096] fp32
                      const bf16*  __restrict__ whh,     // [4096 x 1024] bf16
                      const bf16*  __restrict__ hin,     // [256 x 1024] bf16
                      bf16*        __restrict__ hout,    // [256 x 1024] bf16
                      float*       __restrict__ cst) {   // [256 x 1024] fp32
  __shared__ __align__(16) bf16 lds_a[64 * STRIDE_S];    // ~33.8 KB
  const int tid  = threadIdx.x;
  const int wave = tid >> 5;
  const int lane = tid & 31;
  const int nl   = lane & 15;
  const int hi   = lane >> 4;
  const int mbase0 = blockIdx.y * 64;
  const int jbase  = blockIdx.x * 32 + (wave & 1) * 16;
  const int mw     = (wave >> 1) * 32;       // wave's m-offset in block: 0 or 32

  v8f a00 = {}, a01 = {}, a02 = {}, a03 = {};   // m-position 0, gates i,f,g,o
  v8f a10 = {}, a11 = {}, a12 = {}, a13 = {};   // m-position 1
  const bf16* w0 = whh + (size_t)(0 * HID + jbase + nl) * HID;
  const bf16* w1 = whh + (size_t)(1 * HID + jbase + nl) * HID;
  const bf16* w2 = whh + (size_t)(2 * HID + jbase + nl) * HID;
  const bf16* w3 = whh + (size_t)(3 * HID + jbase + nl) * HID;
  const int arow0 = mw + nl;
  const int arow1 = mw + 16 + nl;
  const int aoff  = hi * 8;

  for (int p = 0; p < 4; ++p) {
    const int kp = p * KPH_S;
    if (p) __syncthreads();
    // async stage 64 rows x 256 K: 2048 16B-chunks / 128 threads = 16 each
#pragma unroll
    for (int i = 0; i < 16; ++i) {
      const int c   = tid + i * 128;
      const int row = c >> 5;               // 32 chunks per row
      const int col = (c & 31) * 8;
      const unsigned ldsoff = (unsigned)(uintptr_t)&lds_a[row * STRIDE_S + col];
      const bf16* g = &hin[(size_t)(mbase0 + row) * HID + kp + col];
      asm volatile("global_load_async_to_lds_b128 %0, %1, off"
                   :: "v"(ldsoff), "v"(g) : "memory");
    }
    asm volatile("s_wait_asynccnt 0x0" ::: "memory");
    __syncthreads();

    for (int k0 = 0; k0 < KPH_S; k0 += 32) {
      v16bf af0, af1;
      ((v8bf*)&af0)[0] = *(const v8bf*)&lds_a[arow0 * STRIDE_S + k0 + aoff];
      ((v8bf*)&af0)[1] = *(const v8bf*)&lds_a[arow0 * STRIDE_S + k0 + aoff + 16];
      ((v8bf*)&af1)[0] = *(const v8bf*)&lds_a[arow1 * STRIDE_S + k0 + aoff];
      ((v8bf*)&af1)[1] = *(const v8bf*)&lds_a[arow1 * STRIDE_S + k0 + aoff + 16];
      const int kb = kp + k0 + hi * 8;
      v16bf b0f, b1f, b2f, b3f;
      ((v8bf*)&b0f)[0] = *(const v8bf*)(w0 + kb);
      ((v8bf*)&b0f)[1] = *(const v8bf*)(w0 + kb + 16);
      ((v8bf*)&b1f)[0] = *(const v8bf*)(w1 + kb);
      ((v8bf*)&b1f)[1] = *(const v8bf*)(w1 + kb + 16);
      ((v8bf*)&b2f)[0] = *(const v8bf*)(w2 + kb);
      ((v8bf*)&b2f)[1] = *(const v8bf*)(w2 + kb + 16);
      ((v8bf*)&b3f)[0] = *(const v8bf*)(w3 + kb);
      ((v8bf*)&b3f)[1] = *(const v8bf*)(w3 + kb + 16);
      __builtin_prefetch(w0 + kb + 64, 0, 3);
      __builtin_prefetch(w1 + kb + 64, 0, 3);
      __builtin_prefetch(w2 + kb + 64, 0, 3);
      __builtin_prefetch(w3 + kb + 64, 0, 3);
      a00 = __builtin_amdgcn_wmma_f32_16x16x32_bf16(false, af0, false, b0f, (short)0, a00, false, false);
      a10 = __builtin_amdgcn_wmma_f32_16x16x32_bf16(false, af1, false, b0f, (short)0, a10, false, false);
      a01 = __builtin_amdgcn_wmma_f32_16x16x32_bf16(false, af0, false, b1f, (short)0, a01, false, false);
      a11 = __builtin_amdgcn_wmma_f32_16x16x32_bf16(false, af1, false, b1f, (short)0, a11, false, false);
      a02 = __builtin_amdgcn_wmma_f32_16x16x32_bf16(false, af0, false, b2f, (short)0, a02, false, false);
      a12 = __builtin_amdgcn_wmma_f32_16x16x32_bf16(false, af1, false, b2f, (short)0, a12, false, false);
      a03 = __builtin_amdgcn_wmma_f32_16x16x32_bf16(false, af0, false, b3f, (short)0, a03, false, false);
      a13 = __builtin_amdgcn_wmma_f32_16x16x32_bf16(false, af1, false, b3f, (short)0, a13, false, false);
    }
  }

  // pointwise LSTM cell on both m-positions (torch gate order i,f,g,o)
  const int j    = jbase + nl;
  const int mloc = hi * 8;
#pragma unroll
  for (int v = 0; v < 8; ++v) {
    {
      const int m = mbase0 + mw + mloc + v;
      const float* xp = xproj + (size_t)m * (4 * HID);
      const float gi = sigf(xp[j] + a00[v]);
      const float gf = sigf(xp[HID + j] + a01[v]);
      const float gg = tanhf(xp[2 * HID + j] + a02[v]);
      const float go = sigf(xp[3 * HID + j] + a03[v]);
      const size_t idx = (size_t)m * HID + j;
      const float cn = gf * cst[idx] + gi * gg;
      cst[idx]  = cn;
      hout[idx] = (bf16)(go * tanhf(cn));
    }
    {
      const int m = mbase0 + mw + 16 + mloc + v;
      const float* xp = xproj + (size_t)m * (4 * HID);
      const float gi = sigf(xp[j] + a10[v]);
      const float gf = sigf(xp[HID + j] + a11[v]);
      const float gg = tanhf(xp[2 * HID + j] + a12[v]);
      const float go = sigf(xp[3 * HID + j] + a13[v]);
      const size_t idx = (size_t)m * HID + j;
      const float cn = gf * cst[idx] + gi * gg;
      cst[idx]  = cn;
      hout[idx] = (bf16)(go * tanhf(cn));
    }
  }
}

// ---------------- host-side launch ----------------

extern "C" void kernel_launch(void* const* d_in, const int* in_sizes, int n_in,
                              void* d_out, int out_size, void* d_ws, size_t ws_size,
                              hipStream_t stream) {
  const float* Cin  = (const float*)d_in[0];  // [256,1024]
  const float* Wih  = (const float*)d_in[1];  // [4096,1024]
  const float* Whh  = (const float*)d_in[2];  // [4096,1024]
  const float* bih  = (const float*)d_in[3];  // [4096]
  const float* bhh  = (const float*)d_in[4];  // [4096]
  const float* Wlin = (const float*)d_in[5];  // [512,1024]
  const float* blin = (const float*)d_in[6];  // [512]
  float* out = (float*)d_out;                 // [128,256,512]
  (void)in_sizes; (void)n_in; (void)out_size; (void)ws_size;

  char* ws = (char*)d_ws;
  size_t off = 0;
  auto carve = [&](size_t bytes) -> char* {
    char* p = ws + off;
    off = (off + bytes + 255) & ~(size_t)255;
    return p;
  };
  bf16*  whh_bf  = (bf16*)carve(4ull * HID * HID * sizeof(bf16));   // 8 MB
  bf16*  wih_bf  = (bf16*)carve(4ull * HID * HID * sizeof(bf16));   // 8 MB
  bf16*  wlin_bf = (bf16*)carve((size_t)OUTD * HID * sizeof(bf16)); // 1 MB
  bf16*  cin_bf  = (bf16*)carve((size_t)BATCH * HID * sizeof(bf16));
  float* bsum    = (float*)carve(4ull * HID * sizeof(float));
  float* xproj   = (float*)carve((size_t)BATCH * 4 * HID * sizeof(float)); // 4 MB
  float* cstate  = (float*)carve((size_t)BATCH * HID * sizeof(float));
  bf16*  hbuf0   = (bf16*)carve((size_t)BATCH * HID * sizeof(bf16));
  bf16*  hbuf1   = (bf16*)carve((size_t)BATCH * HID * sizeof(bf16));

  cvt_kernel<<<1024, 256, 0, stream>>>(Whh, whh_bf, 4 * HID * HID);
  cvt_kernel<<<1024, 256, 0, stream>>>(Wih, wih_bf, 4 * HID * HID);
  cvt_kernel<<<256, 256, 0, stream>>>(Wlin, wlin_bf, OUTD * HID);
  cvt_kernel<<<128, 256, 0, stream>>>(Cin, cin_bf, BATCH * HID);
  bias_sum_kernel<<<16, 256, 0, stream>>>(bih, bhh, bsum, 4 * HID);
  zero_state_kernel<<<512, 256, 0, stream>>>(cstate, hbuf0, BATCH * HID, BATCH * HID);

  // x_proj = C @ W_ih^T + (b_ih + b_hh), computed once (input constant across T)
  gemm_bf16_kernel<<<dim3(4 * HID / 64, BATCH / 32), 256, 0, stream>>>(
      cin_bf, wih_bf, bsum, xproj, 4 * HID);

  bf16* hb[2] = {hbuf0, hbuf1};
  for (int t = 0; t < SEQ; ++t) {
    lstm_step_kernel<<<dim3(HID / 32, BATCH / 64), 128, 0, stream>>>(
        xproj, whh_bf, hb[t & 1], hb[(t + 1) & 1], cstate);
    gemm_bf16_kernel<<<dim3(OUTD / 64, BATCH / 32), 256, 0, stream>>>(
        hb[(t + 1) & 1], wlin_bf, blin, out + (size_t)t * BATCH * OUTD, OUTD);
  }
}
